// Repeat_recommendation_decoder_69861938037643
// MI455X (gfx1250) — matrix operand order repaired and checked by reading, not compiled, not run on hardware
//
#include <hip/hip_runtime.h>
#include <hip/hip_bf16.h>
#include <math.h>

typedef __attribute__((ext_vector_type(16))) __bf16 v16bf;
typedef __attribute__((ext_vector_type(8)))  float  v8f;

#define B_ 256
#define S_ 200
#define H_ 256
#define V_ 50000

// ---------------------------------------------------------------------------
// Kernel 0: zero the [B,V] output (harness poisons it; we accumulate into it)
// ---------------------------------------------------------------------------
__global__ void zero_out_kernel(float* __restrict__ out, int n4) {
    int i = blockIdx.x * blockDim.x + threadIdx.x;
    if (i < n4) {
        float4 z = {0.f, 0.f, 0.f, 0.f};
        ((float4*)out)[i] = z;
    }
}

// ---------------------------------------------------------------------------
// Kernel 1: q[b,k] = sum_h last_memory[b,h] * Wr[k,h]   (tiny, exact fp32)
// ---------------------------------------------------------------------------
__global__ __launch_bounds__(256) void q_kernel(const float* __restrict__ last,
                                                const float* __restrict__ Wr,
                                                float* __restrict__ q) {
    int idx = blockIdx.x * blockDim.x + threadIdx.x;   // b*H + k
    int b = idx >> 8;
    int k = idx & 255;
    const float4* lrow = (const float4*)(last + (size_t)b * H_);
    const float4* wrow = (const float4*)(Wr   + (size_t)k * H_);
    float acc = 0.f;
    #pragma unroll 4
    for (int h = 0; h < H_ / 4; ++h) {
        float4 a = lrow[h], w = wrow[h];
        acc = fmaf(a.x, w.x, acc);
        acc = fmaf(a.y, w.y, acc);
        acc = fmaf(a.z, w.z, acc);
        acc = fmaf(a.w, w.w, acc);
    }
    q[idx] = acc;
}

// ---------------------------------------------------------------------------
// Kernel 2 helpers
// ---------------------------------------------------------------------------
__device__ __forceinline__ void cvt4(v16bf& d, int base, const float4 f) {
    d[base + 0] = (__bf16)f.x;
    d[base + 1] = (__bf16)f.y;
    d[base + 2] = (__bf16)f.z;
    d[base + 3] = (__bf16)f.w;
}

// One 16x16 output tile: A rows = s-tile, B cols = n-tile, K = 256.
// TAIL: last s-tile (rows 192..207; only 192..199 valid).
template <bool TAIL>
__device__ __forceinline__ void process_mt(
    int mt, const float* __restrict__ Abase, const v16bf* __restrict__ Bf,
    float qv, float vv, float* __restrict__ scores, int lh, int hi)
{
    int srow = mt * 16 + lh;
    if (TAIL && srow >= S_) srow = S_ - 1;   // clamp: garbage rows are discarded below
    const float* arow = Abase + (size_t)srow * H_;

    v8f acc = {};
    #pragma unroll
    for (int ks = 0; ks < 8; ++ks) {
        // A layout (16-bit, 16x32): lane supplies row M = lane%16;
        // halves 0..7 -> K = k0..k0+7, halves 8..15 -> K = k0+16..k0+23,
        // with k0 = ks*32 + (lane>=16 ? 8 : 0).
        const int k0 = ks * 32 + hi * 8;
        float4 a0 = *(const float4*)(arow + k0);
        float4 a1 = *(const float4*)(arow + k0 + 4);
        float4 a2 = *(const float4*)(arow + k0 + 16);
        float4 a3 = *(const float4*)(arow + k0 + 20);
        v16bf Af;
        cvt4(Af, 0, a0); cvt4(Af, 4, a1); cvt4(Af, 8, a2); cvt4(Af, 12, a3);
        acc = __builtin_amdgcn_wmma_f32_16x16x32_bf16(
                  false, Af, false, Bf[ks], (short)0, acc, false, false);
    }

    // C layout: VGPR r holds M=r (lanes 0-15) / M=r+8 (lanes 16-31); N = lane%16.
    // scores[s] += sum_n Vr[n] * tanh(q[n] + C[m][n]) : reduce over the 16 lanes
    // of this half-wave (different n), then one LDS atomic per C row.
    #pragma unroll
    for (int r = 0; r < 8; ++r) {
        const int s = mt * 16 + r + hi * 8;
        float c = vv * tanhf(qv + acc[r]);
        c += __shfl_xor(c, 1, 16);
        c += __shfl_xor(c, 2, 16);
        c += __shfl_xor(c, 4, 16);
        c += __shfl_xor(c, 8, 16);
        if (lh == 0 && (!TAIL || s < S_))
            atomicAdd(&scores[s], c);        // ds_add_f32, cross-wave reduce over n-tiles
    }
}

// ---------------------------------------------------------------------------
// Kernel 2: per-batch  scores = Vr . tanh(q + all_memory @ Ur^T)  -> softmax
//           -> scatter-add into out[b, seq_item[b,s]]
// One workgroup (8 wave32) per batch. WMMA bf16 16x16x32, f32 accumulate.
// ---------------------------------------------------------------------------
__global__ __launch_bounds__(256) void attn_scatter_kernel(
    const float*         __restrict__ all_memory,  // [B,S,H]
    const float*         __restrict__ Ur,          // [H,H] (row k_out, col h)
    const float*         __restrict__ Vr,          // [H]
    const float*         __restrict__ q,           // [B,H]
    const unsigned char* __restrict__ mask,        // [B,S] bool
    const int*           __restrict__ seq_item,    // [B,S]
    float*               __restrict__ out)         // [B,V]
{
    __shared__ float scores[S_ + 8];
    __shared__ float red[256];

    const int b    = blockIdx.x;
    const int tid  = threadIdx.x;
    const int wave = tid >> 5;
    const int lane = tid & 31;
    const int lh   = lane & 15;   // lane % 16
    const int hi   = lane >> 4;   // 0 for lanes 0-15, 1 for lanes 16-31

    if (tid < S_ + 8) scores[tid] = 0.f;
    __syncthreads();

    const float* Abase = all_memory + (size_t)b * S_ * H_;

    // Each wave owns 2 of the 16 N-tiles (H output dim)
    #pragma unroll
    for (int nti = 0; nti < 2; ++nti) {
        const int nt       = wave * 2 + nti;
        const int n_global = nt * 16 + lh;

        // Preload all 8 B-fragments for this N-tile (K=256 in steps of 32).
        // B layout (16-bit, 32x16): lane holds N = lane%16; halves 0..7 -> K = k0..k0+7,
        // halves 8..15 -> K = k0+16..k0+23, k0 = kbase + (lane<16 ? 0 : 8).  B[K][N] = Ur[N][K].
        v16bf Bf[8];
        #pragma unroll
        for (int ks = 0; ks < 8; ++ks) {
            const float* brow = Ur + (size_t)n_global * H_ + ks * 32 + hi * 8;
            float4 b0 = *(const float4*)(brow);
            float4 b1 = *(const float4*)(brow + 4);
            float4 b2 = *(const float4*)(brow + 16);
            float4 b3 = *(const float4*)(brow + 20);
            cvt4(Bf[ks], 0, b0); cvt4(Bf[ks], 4, b1);
            cvt4(Bf[ks], 8, b2); cvt4(Bf[ks], 12, b3);
        }

        const float qv = q[b * H_ + n_global];
        const float vv = Vr[n_global];

        for (int mt = 0; mt < 12; ++mt)
            process_mt<false>(mt, Abase, Bf, qv, vv, scores, lh, hi);
        process_mt<true>(12, Abase, Bf, qv, vv, scores, lh, hi);
    }
    __syncthreads();

    // ---------------- masked softmax over S, then scatter ----------------
    float sc = -1e30f;
    if (tid < S_) {
        sc = scores[tid];
        if (mask[(size_t)b * S_ + tid]) sc = -1e9f;
    }
    red[tid] = sc;
    __syncthreads();
    #pragma unroll
    for (int off = 128; off > 0; off >>= 1) {
        if (tid < off) red[tid] = fmaxf(red[tid], red[tid + off]);
        __syncthreads();
    }
    const float mx = red[0];
    __syncthreads();

    float e = 0.f;
    if (tid < S_) e = __expf(sc - mx);
    red[tid] = e;
    __syncthreads();
    #pragma unroll
    for (int off = 128; off > 0; off >>= 1) {
        if (tid < off) red[tid] += red[tid + off];
        __syncthreads();
    }
    const float denom = red[0];

    if (tid < S_) {
        float p = e / denom;
        int item = seq_item[(size_t)b * S_ + tid];
        atomicAdd(&out[(size_t)b * V_ + item], p);   // global_atomic_add_f32
    }
}

// ---------------------------------------------------------------------------
extern "C" void kernel_launch(void* const* d_in, const int* in_sizes, int n_in,
                              void* d_out, int out_size, void* d_ws, size_t ws_size,
                              hipStream_t stream) {
    (void)in_sizes; (void)n_in; (void)out_size; (void)ws_size;

    const int*           seq_item    = (const int*)d_in[0];           // [B,S]
    const float*         last_memory = (const float*)d_in[1];         // [B,H]
    const float*         all_memory  = (const float*)d_in[2];         // [B,S,H]
    const unsigned char* mask        = (const unsigned char*)d_in[3]; // [B,S] bool
    // d_in[4] item_matrix: only its shape is used by the reference
    const float*         Wr          = (const float*)d_in[5];         // [H,H]
    const float*         Ur          = (const float*)d_in[6];         // [H,H]
    const float*         Vr          = (const float*)d_in[7];         // [H]

    float* out = (float*)d_out;            // [B,V]
    float* q   = (float*)d_ws;             // [B,H] scratch (256 KB)

    // 1) zero output (12.8M floats, float4-vectorized)
    const int n4 = (B_ * V_) / 4;
    zero_out_kernel<<<(n4 + 255) / 256, 256, 0, stream>>>(out, n4);

    // 2) q = last_memory @ Wr^T
    q_kernel<<<(B_ * H_) / 256, 256, 0, stream>>>(last_memory, Wr, q);

    // 3) attention scores (WMMA bf16), softmax, scatter-add
    attn_scatter_kernel<<<B_, 256, 0, stream>>>(all_memory, Ur, Vr, q,
                                                mask, seq_item, out);
}
